// SFBlock_73418170958461
// MI455X (gfx1250) — matrix operand order repaired and checked by reading, not compiled, not run on hardware
//
#include <hip/hip_runtime.h>
#include <hip/hip_bf16.h>
#include <math.h>

// ---------------- problem constants ----------------
constexpr int kC      = 128;
constexpr int kHW     = 4096;           // 64*64
constexpr int kB      = 2;
constexpr int kP      = kB * kHW;       // 8192 pixels total
constexpr int kNH     = 4;
constexpr int kHD     = 32;             // head dim
constexpr int kHID    = 256;            // MLP hidden
constexpr int kCPG    = 4;              // channels per group (128/32)

typedef unsigned int u32;
typedef __attribute__((ext_vector_type(16))) __bf16 v16bf;
typedef __attribute__((ext_vector_type(8)))  float  v8f;
typedef __attribute__((ext_vector_type(8)))  int    v8i;

__device__ inline v8f wmma_bf16(v16bf a, v16bf b, v8f c) {
  return __builtin_amdgcn_wmma_f32_16x16x32_bf16(
      false, a, false, b, (short)0, c, false, false);
}

__device__ inline u32 pack2(float lo, float hi) {
  unsigned short l = __builtin_bit_cast(unsigned short, (__bf16)lo);
  unsigned short h = __builtin_bit_cast(unsigned short, (__bf16)hi);
  return (u32)l | ((u32)h << 16);
}

// cross-half (lane ^ 16) exchange via v_permlanex16_b32 (VALU, no LDS)
__device__ inline u32 xhalf_u32(u32 x) {
  return __builtin_amdgcn_permlanex16(x, x, 0x76543210u, 0xfedcba98u, false, false);
}
__device__ inline float xhalf_f32(float x) {
  return __builtin_bit_cast(float, xhalf_u32(__builtin_bit_cast(u32, x)));
}

// ---------- WMMA fragment loaders; all elements are packed bf16-pair dwords ----------
// A 16x32 (MxK) from paired weights u32[M][K/2]
__device__ inline v16bf load_a_pk(const u32* __restrict__ p, int ldp, int lane) {
  int m  = lane & 15;
  int kh = (lane & 16) ? 4 : 0;
  v8i t;
#pragma unroll
  for (int v = 0; v < 8; ++v) {
    int kp2 = v + 2 * kh;
    t[v] = (int)p[m * ldp + kp2];
  }
  return __builtin_bit_cast(v16bf, t);
}
// A fragment from k-paired source where elem-pair (m, kp2) sits at p[kp2*ld + m]
__device__ inline v16bf load_a_t_pk(const u32* __restrict__ p, int ld, int lane) {
  int m  = lane & 15;
  int kh = (lane & 16) ? 4 : 0;
  v8i t;
#pragma unroll
  for (int v = 0; v < 8; ++v) {
    int kp2 = v + 2 * kh;
    t[v] = (int)p[(size_t)kp2 * ld + m];
  }
  return __builtin_bit_cast(v16bf, t);
}
// A fragment from plain bf16 rows: elem(m,k) = bp[m*ld + k] (k contiguous -> dwords)
__device__ inline v16bf load_a_row_pk(const __bf16* __restrict__ bp, int ld, int lane) {
  int m  = lane & 15;
  int kh = (lane & 16) ? 4 : 0;
  const u32* p = (const u32*)bp;           // all offsets even
  v8i t;
#pragma unroll
  for (int v = 0; v < 8; ++v) {
    int kp2 = v + 2 * kh;
    t[v] = (int)p[((size_t)m * ld >> 1) + kp2];
  }
  return __builtin_bit_cast(v16bf, t);
}
// B 32x16 (KxN) from paired activations u32[K/2][P]: elem-pair (kp, n) at p[kp*ld + n]
__device__ inline v16bf load_b_pk(const u32* __restrict__ p, int ld, int lane) {
  int n  = lane & 15;
  int kh = (lane & 16) ? 8 : 0;
  v8i t;
#pragma unroll
  for (int v = 0; v < 8; ++v) t[v] = (int)p[(size_t)(kh + v) * ld + n];
  return __builtin_bit_cast(v16bf, t);
}

// ---------------- GroupNorm: per-(b,group) mean / rsqrt(var) ----------------
__global__ __launch_bounds__(256)
void gn_stats_kernel(const float* __restrict__ x, float* __restrict__ stats) {
  int bg = blockIdx.x;                       // b*32+g
  int b = bg >> 5, g = bg & 31;
  const float* base = x + (size_t)b * kC * kHW + (size_t)g * kCPG * kHW;
  float s = 0.f, ss = 0.f;
  for (int i = threadIdx.x; i < kCPG * kHW; i += 256) {
    float v = base[i];
    s += v; ss += v * v;
  }
  __shared__ float red0[256], red1[256];
  red0[threadIdx.x] = s; red1[threadIdx.x] = ss;
  __syncthreads();
  for (int off = 128; off > 0; off >>= 1) {
    if (threadIdx.x < off) {
      red0[threadIdx.x] += red0[threadIdx.x + off];
      red1[threadIdx.x] += red1[threadIdx.x + off];
    }
    __syncthreads();
  }
  if (threadIdx.x == 0) {
    const float invN = 1.0f / (kCPG * kHW);
    float mu  = red0[0] * invN;
    float var = red1[0] * invN - mu * mu;
    stats[bg * 2]     = mu;
    stats[bg * 2 + 1] = rsqrtf(var + 1e-5f);
  }
}

// normalize (B,C,H,W) f32 -> channel-pair-packed u32 [C/2][B*HW]
__global__ __launch_bounds__(256)
void gn_apply_kernel(const float* __restrict__ x, const float* __restrict__ stats,
                     const float* __restrict__ gamma, const float* __restrict__ beta,
                     u32* __restrict__ out) {
  int i = blockIdx.x * 256 + threadIdx.x;    // over (kC/2)*kP
  if (i >= (kC / 2) * kP) return;
  int cp = i / kP, p = i - cp * kP;
  int c0 = cp * 2;
  int b = p >> 12, pp = p & (kHW - 1);
  const float* src = x + (size_t)b * kC * kHW + (size_t)c0 * kHW + pp;
  float v0 = src[0], v1 = src[kHW];          // channels c0, c0+1 (same group)
  int sg = (b << 5) + (c0 >> 2);
  float mu = stats[sg * 2], rs = stats[sg * 2 + 1];
  float y0 = (v0 - mu) * rs * gamma[c0]     + beta[c0];
  float y1 = (v1 - mu) * rs * gamma[c0 + 1] + beta[c0 + 1];
  out[i] = pack2(y0, y1);
}

// ---------------- weights f32 -> packed bf16 pairs u32[M][K/2] ----------------
__global__ __launch_bounds__(256)
void wconv_kernel(const float* __restrict__ src, u32* __restrict__ dst,
                  int M, int K, int transpose) {
  int i = blockIdx.x * 256 + threadIdx.x;
  int kp2 = K >> 1;
  if (i >= M * kp2) return;
  int m = i / kp2, kp = i - m * kp2;
  int k0 = kp * 2;
  float a = transpose ? src[k0 * M + m]       : src[m * K + k0];
  float b = transpose ? src[(k0 + 1) * M + m] : src[m * K + k0 + 1];
  dst[i] = pack2(a, b);
}

// ---------------- WMMA GEMM: Out[M, kP] = A[M,K] * X[K, kP] ----------------
// wave computes 64x16 (4 m-tiles share one B fragment per k-step); K templated.
// MODE 0: +bias, channel-paired u32 out      (Q, K projections)
// MODE 1: +bias, plain bf16 out              (V projection)
// MODE 2: +bias, GELU, paired out            (MLP1)
// MODE 3: +bias, +f32 residual, paired out   (MLP2 + skip)
// MODE 4: +bias, f32 out to (B,C,H,W)        (final Wo -> d_out)
template <int K, int MODE>
__global__ __launch_bounds__(128)
void gemm_kernel(const u32* __restrict__ Ap, const u32* __restrict__ Xp,
                 const float* __restrict__ bias, void* __restrict__ outp,
                 const float* __restrict__ resid) {
  constexpr int KP2 = K / 2;
  int wave = threadIdx.x >> 5, lane = threadIdx.x & 31;
  int m0 = blockIdx.y * 64;
  int n0 = blockIdx.x * 64 + wave * 16;
  v8f acc[4] = {{}, {}, {}, {}};
#pragma unroll
  for (int k0 = 0; k0 < K; k0 += 32) {
    v16bf bb = load_b_pk(Xp + (size_t)(k0 / 2) * kP + n0, kP, lane);
#pragma unroll
    for (int t = 0; t < 4; ++t) {
      v16bf a = load_a_pk(Ap + (size_t)(m0 + 16 * t) * KP2 + k0 / 2, KP2, lane);
      acc[t] = wmma_bf16(a, bb, acc[t]);
    }
  }
  int n = n0 + (lane & 15);
#pragma unroll
  for (int t = 0; t < 4; ++t) {
    int mbase = m0 + 16 * t + ((lane >> 4) << 3);   // even
    if constexpr (MODE == 4) {
      int b = n >> 12, pp = n & (kHW - 1);
      float* o = (float*)outp + (size_t)b * kC * kHW + pp;
#pragma unroll
      for (int v = 0; v < 8; ++v)
        o[(size_t)(mbase + v) * kHW] = acc[t][v] + bias[mbase + v];
    } else if constexpr (MODE == 1) {
      __bf16* o = (__bf16*)outp;
#pragma unroll
      for (int v = 0; v < 8; ++v)
        o[(size_t)(mbase + v) * kP + n] = (__bf16)(acc[t][v] + bias[mbase + v]);
    } else {
      u32* o = (u32*)outp;
#pragma unroll
      for (int v = 0; v < 8; v += 2) {
        float a0 = acc[t][v]     + bias[mbase + v];
        float a1 = acc[t][v + 1] + bias[mbase + v + 1];
        if constexpr (MODE == 2) {
          a0 = 0.5f * a0 * (1.0f + erff(a0 * 0.70710678118654752f));
          a1 = 0.5f * a1 * (1.0f + erff(a1 * 0.70710678118654752f));
        } else if constexpr (MODE == 3) {
          a0 += resid[(size_t)(mbase + v) * kP + n];
          a1 += resid[(size_t)(mbase + v + 1) * kP + n];
        }
        o[(size_t)((mbase + v) >> 1) * kP + n] = pack2(a0, a1);
      }
    }
  }
}

// ---------------- flash attention, transposed (per (b, head)) ----------------
// S^T[kpos,q] = K·Q^T  and  O^T[d,q] = V^T·P^T: the softmax reduction direction
// (kpos) lies within a lane -> in-lane fmax/fadd chains + one permlanex16 combine.
// Qp/Kp: channel-paired u32 [C/2][kP]; Vb: plain bf16 [C][kP].
__global__ __launch_bounds__(128)
void flash_attn_kernel(const u32* __restrict__ Qp, const u32* __restrict__ Kp,
                       const __bf16* __restrict__ Vb, float* __restrict__ attnF,
                       u32* __restrict__ attnP) {
  int wave = threadIdx.x >> 5, lane = threadIdx.x & 31;
  int b = blockIdx.z, h = blockIdx.y;
  int qbase   = blockIdx.x * 64 + wave * 16;
  int colbase = b * kHW;
  int hch = h * kHD;                         // head channel base (multiple of 32)

  const u32*    qp  = Qp + (size_t)(hch >> 1) * kP + colbase + qbase;
  const u32*    kpn = Kp + (size_t)(hch >> 1) * kP + colbase;
  const __bf16* vp  = Vb + (size_t)hch * kP + colbase;

  // Q as the B-matrix (k = head-dim, n = q): loaded once per wave
  v16bf bq = load_b_pk(qp, kP, lane);

  v8f o0 = {}, o1 = {};                      // O^T rows d0..15 / d16..31, cols q
  float mi = -1e30f, li = 0.f;               // per-lane (= per-q) running stats
  const float scale = 0.17677669529663687f;  // 1/sqrt(32)
  bool lolane = (lane & 16) == 0;

  for (int kb = 0; kb < kHW; kb += 32) {
    // S^T tiles: rows kpos kb..kb+15 and kb+16..kb+31, cols q
    v16bf ak0 = load_a_t_pk(kpn + kb, kP, lane);
    v16bf ak1 = load_a_t_pk(kpn + kb + 16, kP, lane);
    v8f st0 = {}, st1 = {};
    st0 = wmma_bf16(ak0, bq, st0);
    st1 = wmma_bf16(ak1, bq, st1);

    if (kb + 32 < kHW) {                     // global_prefetch_b8 on next tiles
      __builtin_prefetch(kpn + kb + 32, 0, 1);
      __builtin_prefetch(vp + kb + 32, 0, 1);
    }

    // online softmax: 16 in-lane values + one cross-half combine
    float sx0[8], sx1[8];
    float mx = -1e30f;
#pragma unroll
    for (int v = 0; v < 8; ++v) {
      sx0[v] = st0[v] * scale;
      sx1[v] = st1[v] * scale;
      mx = fmaxf(mx, fmaxf(sx0[v], sx1[v]));
    }
    mx = fmaxf(mx, xhalf_f32(mx));
    float mn = fmaxf(mi, mx);
    float alpha = __expf(mi - mn);
    mi = mn;
    float e0[8], e1[8];
    float r = 0.f;
#pragma unroll
    for (int v = 0; v < 8; ++v) {
      e0[v] = __expf(sx0[v] - mn);
      e1[v] = __expf(sx1[v] - mn);
      r += e0[v] + e1[v];
    }
    r = r + xhalf_f32(r);
    li = li * alpha + r;
#pragma unroll
    for (int v = 0; v < 8; ++v) { o0[v] *= alpha; o1[v] *= alpha; }

    // assemble P^T as a B fragment: in-lane bf16 pair packing + 4 half-exchanges
    v8i tb;
#pragma unroll
    for (int j = 0; j < 4; ++j) {
      u32 own0 = pack2(e0[2 * j], e0[2 * j + 1]);   // kpos (8H)+2j,2j+1
      u32 own1 = pack2(e1[2 * j], e1[2 * j + 1]);   // kpos 16+(8H)+2j,2j+1
      u32 send = lolane ? own1 : own0;
      u32 recv = xhalf_u32(send);
      tb[j]     = lolane ? (int)own0 : (int)recv;
      tb[j + 4] = lolane ? (int)recv : (int)own1;
    }
    v16bf bp = __builtin_bit_cast(v16bf, tb);

    // O^T += V^T * P^T (V pairs contiguous along pixels)
    v16bf av0 = load_a_row_pk(vp + kb, kP, lane);                     // d 0..15
    v16bf av1 = load_a_row_pk(vp + (size_t)16 * kP + kb, kP, lane);   // d 16..31
    o0 = wmma_bf16(av0, bp, o0);
    o1 = wmma_bf16(av1, bp, o1);
  }

  // epilogue: divide by li; channel pairs are in-lane -> direct paired stores
  float inv = 1.0f / li;
  int q  = qbase + (lane & 15);
  int db = (lane & 16) ? 8 : 0;
#pragma unroll
  for (int v = 0; v < 8; ++v) {
    float r0 = o0[v] * inv, r1 = o1[v] * inv;
    attnF[(size_t)(hch + db + v) * kP + colbase + q]      = r0;
    attnF[(size_t)(hch + 16 + db + v) * kP + colbase + q] = r1;
  }
#pragma unroll
  for (int v = 0; v < 8; v += 2) {
    attnP[(size_t)((hch + db + v) >> 1) * kP + colbase + q] =
        pack2(o0[v] * inv, o0[v + 1] * inv);
    attnP[(size_t)((hch + 16 + db + v) >> 1) * kP + colbase + q] =
        pack2(o1[v] * inv, o1[v + 1] * inv);
  }
}

// ---------------- host-side orchestration ----------------
extern "C" void kernel_launch(void* const* d_in, const int* in_sizes, int n_in,
                              void* d_out, int out_size, void* d_ws, size_t ws_size,
                              hipStream_t stream) {
  (void)in_sizes; (void)n_in; (void)out_size; (void)ws_size;
  const float* feat_opt = (const float*)d_in[0];
  const float* feat_sar = (const float*)d_in[1];
  const float* gq  = (const float*)d_in[2];  const float* bq  = (const float*)d_in[3];
  const float* gkv = (const float*)d_in[4];  const float* bkv = (const float*)d_in[5];
  const float* Wq  = (const float*)d_in[6];  const float* bqp = (const float*)d_in[7];
  const float* Wk  = (const float*)d_in[8];  const float* bkp = (const float*)d_in[9];
  const float* Wv  = (const float*)d_in[10]; const float* bvp = (const float*)d_in[11];
  const float* W1  = (const float*)d_in[12]; const float* b1  = (const float*)d_in[13];
  const float* W2  = (const float*)d_in[14]; const float* b2  = (const float*)d_in[15];
  const float* Wo  = (const float*)d_in[16]; const float* bo  = (const float*)d_in[17];

  char* ws = (char*)d_ws;
  auto carve = [&](size_t bytes) -> char* {
    char* p = ws; ws += (bytes + 255) & ~(size_t)255; return p;
  };
  float* statsQ  = (float*)carve(64 * 2 * sizeof(float));
  float* statsKV = (float*)carve(64 * 2 * sizeof(float));
  u32* XqP    = (u32*)carve((size_t)(kC / 2) * kP * 4);
  u32* XkvP   = (u32*)carve((size_t)(kC / 2) * kP * 4);
  u32* WqP    = (u32*)carve((size_t)kC * (kC / 2) * 4);
  u32* WkP    = (u32*)carve((size_t)kC * (kC / 2) * 4);
  u32* WvP    = (u32*)carve((size_t)kC * (kC / 2) * 4);
  u32* WoP    = (u32*)carve((size_t)kC * (kC / 2) * 4);
  u32* A1P    = (u32*)carve((size_t)kHID * (kC / 2) * 4);   // W1^T [hid][C/2]
  u32* A2P    = (u32*)carve((size_t)kC * (kHID / 2) * 4);   // W2^T [C][hid/2]
  u32* QP     = (u32*)carve((size_t)(kC / 2) * kP * 4);
  u32* KP     = (u32*)carve((size_t)(kC / 2) * kP * 4);
  __bf16* Vb  = (__bf16*)carve((size_t)kC * kP * 2);
  float* attnF = (float*)carve((size_t)kC * kP * 4);
  u32* attnP  = (u32*)carve((size_t)(kC / 2) * kP * 4);
  u32* GP     = (u32*)carve((size_t)(kHID / 2) * kP * 4);
  u32* fusedP = (u32*)carve((size_t)(kC / 2) * kP * 4);

  dim3 blk(128);

  // 1) GroupNorm stats + normalize to channel-paired bf16
  gn_stats_kernel<<<64, 256, 0, stream>>>(feat_opt, statsQ);
  gn_stats_kernel<<<64, 256, 0, stream>>>(feat_sar, statsKV);
  int napply = ((kC / 2) * kP + 255) / 256;
  gn_apply_kernel<<<napply, 256, 0, stream>>>(feat_opt, statsQ, gq, bq, XqP);
  gn_apply_kernel<<<napply, 256, 0, stream>>>(feat_sar, statsKV, gkv, bkv, XkvP);

  // 2) weights -> packed bf16 pairs (W1/W2 transposed so A is row-major [M][K/2])
  wconv_kernel<<<32, 256, 0, stream>>>(Wq, WqP, kC, kC, 0);
  wconv_kernel<<<32, 256, 0, stream>>>(Wk, WkP, kC, kC, 0);
  wconv_kernel<<<32, 256, 0, stream>>>(Wv, WvP, kC, kC, 0);
  wconv_kernel<<<32, 256, 0, stream>>>(Wo, WoP, kC, kC, 0);
  wconv_kernel<<<64, 256, 0, stream>>>(W1, A1P, kHID, kC, 1);
  wconv_kernel<<<64, 256, 0, stream>>>(W2, A2P, kC, kHID, 1);

  // 3) Q/K/V projections (M=128, K=128, N=8192)
  dim3 g128(kP / 64, kC / 64);
  gemm_kernel<128, 0><<<g128, blk, 0, stream>>>(WqP, XqP,  bqp, QP, nullptr);
  gemm_kernel<128, 0><<<g128, blk, 0, stream>>>(WkP, XkvP, bkp, KP, nullptr);
  gemm_kernel<128, 1><<<g128, blk, 0, stream>>>(WvP, XkvP, bvp, Vb, nullptr);

  // 4) flash attention: grid = (HW/64 q-blocks, heads, batch)
  dim3 gfa(kHW / 64, kNH, kB);
  flash_attn_kernel<<<gfa, blk, 0, stream>>>(QP, KP, Vb, attnF, attnP);

  // 5) MLP1 (GELU), MLP2 (+residual), final Wo -> d_out (f32, BCHW)
  dim3 g256(kP / 64, kHID / 64);
  gemm_kernel<128, 2><<<g256, blk, 0, stream>>>(A1P, attnP, b1, GP, nullptr);
  gemm_kernel<256, 3><<<g128, blk, 0, stream>>>(A2P, GP, b2, fusedP, attnF);
  gemm_kernel<128, 4><<<g128, blk, 0, stream>>>(WoP, fusedP, bo, d_out, nullptr);
}
